// MultiHeadAttention_29832842838455
// MI455X (gfx1250) — matrix operand order repaired and checked by reading, not compiled, not run on hardware
//
#include <hip/hip_runtime.h>
#include <hip/hip_bf16.h>
#include <math.h>

// ---------------------------------------------------------------------------
// MHA forward for MI455X (gfx1250, wave32, WMMA).
//   NB=4, L=2048, E=1024, H=16, D=64, OUT=1024
// Compute-bound (~137 GFLOP vs <200MB HBM traffic) => route all GEMMs through
// v_wmma_f32_16x16x32_bf16, flash-style attention (no 1GB f32 score tensor), and
// double-buffered GLOBAL_LOAD_ASYNC_TO_LDS_B128 staging: issue tile t+1,
// s_wait_asynccnt <= batch (in-order completion) => copy overlaps WMMA.
// ---------------------------------------------------------------------------

typedef __attribute__((ext_vector_type(16))) __bf16        v16bf;
typedef __attribute__((ext_vector_type(8)))  float         v8f;
typedef __attribute__((ext_vector_type(8)))  unsigned short v8us;

union Frag16 {
  v16bf v;
  v8us  h[2];
};

__device__ __forceinline__ unsigned short f32_to_bf16_rne(float f) {
  unsigned int u = __builtin_bit_cast(unsigned int, f);
  u += 0x7FFFu + ((u >> 16) & 1u);
  return (unsigned short)(u >> 16);
}

__device__ __forceinline__ v8f v8f_zero() {
  v8f z;
#pragma unroll
  for (int i = 0; i < 8; ++i) z[i] = 0.0f;
  return z;
}

__device__ __forceinline__ v8f wmma_bf16(const Frag16& a, const Frag16& b, v8f c) {
  // D = A(16x32 bf16) * B(32x16 bf16) + C(16x16 f32)
  return __builtin_amdgcn_wmma_f32_16x16x32_bf16(
      /*neg_a=*/false, a.v, /*neg_b=*/false, b.v,
      /*c_mod=*/(short)0, c, /*reuse_a=*/false, /*reuse_b=*/false);
}

// Generic pointers to __shared__ carry the LDS byte offset in addr[31:0]
// (ISA aperture scheme), which is exactly what the async-load VDST wants.
__device__ __forceinline__ unsigned lds_off(const void* p) {
  return (unsigned)(size_t)p;
}

// GLOBAL_LOAD_ASYNC_TO_LDS_B128, GVS mode: saddr = uniform base (SGPR pair),
// vaddr = per-lane 32-bit byte offset, vdst = per-lane LDS byte address.
// Tracked by ASYNCcnt; bypasses VGPRs entirely (CDNA5 data-mover path).
__device__ __forceinline__ void async_ld_b128(unsigned lds_addr, unsigned goff,
                                              const void* base) {
  asm volatile("global_load_async_to_lds_b128 %0, %1, %2"
               :: "v"(lds_addr), "v"(goff), "s"(base)
               : "memory");
}

// ---------------------------------------------------------------------------
// Kernel 1: f32 -> bf16 (RNE) grid-stride convert.
// ---------------------------------------------------------------------------
__global__ __launch_bounds__(256) void convert_f32_to_bf16(
    const float* __restrict__ src, unsigned short* __restrict__ dst, size_t n) {
  size_t i = blockIdx.x * (size_t)blockDim.x + threadIdx.x;
  size_t stride = (size_t)gridDim.x * blockDim.x;
  for (; i < n; i += stride) dst[i] = f32_to_bf16_rne(src[i]);
}

// ---------------------------------------------------------------------------
// Kernel 2/4: NT GEMM  C[M,N] = A[M,K] * B[N,K]^T  (both K-contiguous, bf16).
// Block tile 128x64, 8 waves, each wave computes 32x32 (2x2 WMMA tiles), BK=32.
// Double-buffered async-to-LDS staging (3 b128/thread/tile => wait <= 0x3).
// BF16_OUT: write bf16 (QKV).  else: f32 + bias (out-projection -> d_out).
// ---------------------------------------------------------------------------
template <bool BF16_OUT>
__global__ __launch_bounds__(256) void gemm_wmma_nt(
    const unsigned short* __restrict__ A,   // [M][K] bf16
    const unsigned short* __restrict__ B,   // [N][K] bf16 (torch weight layout)
    unsigned short* __restrict__ Cb,        // [M][N] bf16 (if BF16_OUT)
    float* __restrict__ Cf,                 // [M][N] f32 (else)
    const float* __restrict__ bias,         // [N] (else)
    int M, int N, int K) {
  constexpr int BM = 128, BN = 64, BK = 32, LDT = BK + 8;
  __shared__ alignas(16) unsigned short As[2][BM][LDT];
  __shared__ alignas(16) unsigned short Bs[2][BN][LDT];

  const int tid   = threadIdx.x;
  const int wave  = tid >> 5;
  const int lane  = tid & 31;
  const int ln    = lane & 15;   // col (B/C) or row (A) within a 16-wide tile
  const int lh    = lane >> 4;   // half-wave selector per ISA fragment tables
  const int warpM = wave >> 1;   // 0..3
  const int warpN = wave & 1;    // 0..1
  const int bm    = blockIdx.y * BM;
  const int bn    = blockIdx.x * BN;

  // Stage one 128x32 A tile + 64x32 B tile: 3 async b128 per thread.
  auto stage = [&](int t, int buf) {
#pragma unroll
    for (int i = 0; i < 2; ++i) {
      int c = tid + i * 256;
      int row = c >> 2, kc = c & 3;
      unsigned goff = (unsigned)(((size_t)(bm + row) * K + t * BK + kc * 8) *
                                 sizeof(unsigned short));
      async_ld_b128(lds_off(&As[buf][row][kc * 8]), goff, A);
    }
    {
      int row = tid >> 2, kc = tid & 3;
      unsigned goff = (unsigned)(((size_t)(bn + row) * K + t * BK + kc * 8) *
                                 sizeof(unsigned short));
      async_ld_b128(lds_off(&Bs[buf][row][kc * 8]), goff, B);
    }
  };

  v8f acc[2][2];
#pragma unroll
  for (int i = 0; i < 2; ++i)
#pragma unroll
    for (int j = 0; j < 2; ++j) acc[i][j] = v8f_zero();

  const int nT = K / BK;
  stage(0, 0);
  for (int t = 0; t < nT; ++t) {
    const int cur = t & 1;
    if (t + 1 < nT) {
      stage(t + 1, cur ^ 1);
      // 3 new loads outstanding; <=3 means tile t's batch has landed.
      asm volatile("s_wait_asynccnt 0x3" ::: "memory");
    } else {
      asm volatile("s_wait_asynccnt 0x0" ::: "memory");
    }
    __syncthreads();

    // A fragment: lane row = ln, K chunks [8h,8h+8) and [16+8h,16+8h+8).
    Frag16 af[2];
#pragma unroll
    for (int tt = 0; tt < 2; ++tt) {
      int m = warpM * 32 + tt * 16 + ln;
      af[tt].h[0] = *(const v8us*)(&As[cur][m][8 * lh]);
      af[tt].h[1] = *(const v8us*)(&As[cur][m][16 + 8 * lh]);
    }
    // B fragment: lane col = ln, K chunks [8h,8h+8) and [16+8h,16+8h+8).
    Frag16 bf[2];
#pragma unroll
    for (int tt = 0; tt < 2; ++tt) {
      int n = warpN * 32 + tt * 16 + ln;
      bf[tt].h[0] = *(const v8us*)(&Bs[cur][n][8 * lh]);
      bf[tt].h[1] = *(const v8us*)(&Bs[cur][n][16 + 8 * lh]);
    }
#pragma unroll
    for (int i = 0; i < 2; ++i)
#pragma unroll
      for (int j = 0; j < 2; ++j) acc[i][j] = wmma_bf16(af[i], bf[j], acc[i][j]);
    __syncthreads();  // protects buffer reused two iterations later
  }

  // C/D layout: VGPR v -> row (v + 8*lh), col = ln.
#pragma unroll
  for (int i = 0; i < 2; ++i)
#pragma unroll
    for (int j = 0; j < 2; ++j) {
      int gr0 = bm + warpM * 32 + i * 16 + 8 * lh;
      int gc  = bn + warpN * 32 + j * 16 + ln;
#pragma unroll
      for (int v = 0; v < 8; ++v) {
        float val = acc[i][j][v];
        size_t idx = (size_t)(gr0 + v) * N + gc;
        if constexpr (BF16_OUT) {
          Cb[idx] = f32_to_bf16_rne(val);
        } else {
          Cf[idx] = val + bias[gc];
        }
      }
    }
}

// ---------------------------------------------------------------------------
// Kernel 3: flash attention per (batch, head).
// Grid: (L/64, NB*H). 128 threads = 4 waves; each wave owns 16 query rows.
// Key blocks of 32: S = Q*K^T (4 WMMAs), online softmax in registers,
// P staged through per-wave LDS (C-layout -> A-layout), O += P*V (4 WMMAs).
// K tile via double-buffered async-to-LDS (2 b128/thread => wait <= 0x2);
// V tile transposed manually (2B scatter) into the same alternate buffer.
// ---------------------------------------------------------------------------
__global__ __launch_bounds__(128) void flash_attn_wmma(
    const unsigned short* __restrict__ qkv,  // [NB*L][H*192] bf16
    unsigned short* __restrict__ obf) {      // [NB*L][H*64]  bf16
  constexpr int L = 2048, H = 16, D = 64, RS = H * 3 * D;  // row stride 3072
  constexpr int BQ = 64, BK = 32;
  __shared__ alignas(16) unsigned short Kt[2][BK][D + 8];   // [key][d]
  __shared__ alignas(16) unsigned short VtT[2][D][BK + 8];  // [d][key]
  __shared__ alignas(16) unsigned short Pt[4][16][40];      // per-wave P scratch

  const int tid  = threadIdx.x;
  const int w    = tid >> 5;
  const int lane = tid & 31;
  const int ln   = lane & 15;
  const int lh   = lane >> 4;

  const int bh = blockIdx.y;
  const int b  = bh >> 4;
  const int h  = bh & 15;
  const int q0 = blockIdx.x * BQ + w * 16;  // query row base within batch
  const size_t brow = (size_t)b * L;

  // Stage one 32-key K/V block into buffer `buf`.
  auto stage_kv = [&](int kb, int buf) {
    const int key0 = kb * BK;
#pragma unroll
    for (int i = 0; i < 2; ++i) {
      int c = tid + i * 128;
      int key = c >> 3, ch = c & 7;
      unsigned goff = (unsigned)(((brow + key0 + key) * RS + h * (3 * D) + D +
                                  ch * 8) * sizeof(unsigned short));
      async_ld_b128(lds_off(&Kt[buf][key][ch * 8]), goff, qkv);
    }
#pragma unroll
    for (int i = 0; i < 2; ++i) {
      int c = tid + i * 128;
      int key = c >> 3, ch = c & 7;
      const unsigned short* vp =
          qkv + (brow + key0 + key) * RS + h * (3 * D) + 2 * D + ch * 8;
      v8us dv = *(const v8us*)vp;
#pragma unroll
      for (int j = 0; j < 8; ++j) VtT[buf][ch * 8 + j][key] = dv[j];
    }
  };

  // Q A-fragments for the two 32-wide d chunks, kept in registers all kernel.
  Frag16 qf[2];
  {
    const unsigned short* qrow = qkv + (brow + q0 + ln) * RS + h * (3 * D);
#pragma unroll
    for (int kc = 0; kc < 2; ++kc) {
      qf[kc].h[0] = *(const v8us*)(qrow + 32 * kc + 8 * lh);
      qf[kc].h[1] = *(const v8us*)(qrow + 32 * kc + 16 + 8 * lh);
    }
  }

  float mrun[8], lrun[8];
#pragma unroll
  for (int v = 0; v < 8; ++v) { mrun[v] = -3.0e38f; lrun[v] = 0.0f; }
  v8f oacc[4];
#pragma unroll
  for (int dt = 0; dt < 4; ++dt) oacc[dt] = v8f_zero();

  const float scale = 0.125f;  // 1/sqrt(64)
  const int nKB = L / BK;

  stage_kv(0, 0);
  for (int kb = 0; kb < nKB; ++kb) {
    const int cur = kb & 1;
    if (kb + 1 < nKB) {
      stage_kv(kb + 1, cur ^ 1);
      // 2 new async K loads outstanding; <=2 => tile kb's K batch landed.
      asm volatile("s_wait_asynccnt 0x2" ::: "memory");
    } else {
      asm volatile("s_wait_asynccnt 0x0" ::: "memory");
    }
    __syncthreads();

    // S (16 q-rows x 32 keys) = Q * K^T ; contraction over D=64 = 2 WMMA steps.
    v8f s[2];
    s[0] = v8f_zero();
    s[1] = v8f_zero();
#pragma unroll
    for (int nt = 0; nt < 2; ++nt) {
#pragma unroll
      for (int kc = 0; kc < 2; ++kc) {
        Frag16 kf;  // B fragment: col = key (ln), K(d) chunks [32kc+8lh,+8), [32kc+16+8lh,+8)
        int n = nt * 16 + ln;
        kf.h[0] = *(const v8us*)(&Kt[cur][n][32 * kc + 8 * lh]);
        kf.h[1] = *(const v8us*)(&Kt[cur][n][32 * kc + 16 + 8 * lh]);
        s[nt] = wmma_bf16(qf[kc], kf, s[nt]);
      }
    }

    // Online softmax; row r = v + 8*lh replicated across the 16-lane group.
#pragma unroll
    for (int v = 0; v < 8; ++v) {
      float a0 = s[0][v] * scale;
      float a1 = s[1][v] * scale;
      float mx = fmaxf(a0, a1);
      mx = fmaxf(mx, __shfl_xor(mx, 1, 32));
      mx = fmaxf(mx, __shfl_xor(mx, 2, 32));
      mx = fmaxf(mx, __shfl_xor(mx, 4, 32));
      mx = fmaxf(mx, __shfl_xor(mx, 8, 32));
      float mnew  = fmaxf(mrun[v], mx);
      float alpha = __expf(mrun[v] - mnew);
      float p0 = __expf(a0 - mnew);
      float p1 = __expf(a1 - mnew);
      float rs = p0 + p1;
      rs += __shfl_xor(rs, 1, 32);
      rs += __shfl_xor(rs, 2, 32);
      rs += __shfl_xor(rs, 4, 32);
      rs += __shfl_xor(rs, 8, 32);
      lrun[v] = lrun[v] * alpha + rs;
      mrun[v] = mnew;
#pragma unroll
      for (int dt = 0; dt < 4; ++dt) oacc[dt][v] *= alpha;
      // C-layout -> LDS so we can re-read P as an A fragment.
      Pt[w][v + 8 * lh][ln]      = f32_to_bf16_rne(p0);
      Pt[w][v + 8 * lh][16 + ln] = f32_to_bf16_rne(p1);
    }

    // Intra-wave LDS RAW on the per-wave Pt region.
    asm volatile("s_wait_dscnt 0x0" ::: "memory");

    Frag16 pf;  // A fragment of P (16 rows x 32 keys)
    pf.h[0] = *(const v8us*)(&Pt[w][ln][8 * lh]);
    pf.h[1] = *(const v8us*)(&Pt[w][ln][16 + 8 * lh]);
#pragma unroll
    for (int dt = 0; dt < 4; ++dt) {
      Frag16 vf;  // B fragment of V: col = d (ln), K(key) chunks [8lh,+8), [16+8lh,+8)
      int dc = dt * 16 + ln;
      vf.h[0] = *(const v8us*)(&VtT[cur][dc][8 * lh]);
      vf.h[1] = *(const v8us*)(&VtT[cur][dc][16 + 8 * lh]);
      oacc[dt] = wmma_bf16(pf, vf, oacc[dt]);
    }
    __syncthreads();  // buffer `cur` is refilled next iteration
  }

  // Normalize and store O (bf16, layout [token][h*64+d]).
#pragma unroll
  for (int v = 0; v < 8; ++v) {
    float inv = 1.0f / lrun[v];
    size_t r = brow + q0 + v + 8 * lh;
#pragma unroll
    for (int dt = 0; dt < 4; ++dt) {
      int c = h * D + dt * 16 + ln;
      obf[r * (H * D) + c] = f32_to_bf16_rne(oacc[dt][v] * inv);
    }
  }
}

// ---------------------------------------------------------------------------
// Host-side orchestration (all on `stream`, graph-capture safe).
// ---------------------------------------------------------------------------
extern "C" void kernel_launch(void* const* d_in, const int* in_sizes, int n_in,
                              void* d_out, int out_size, void* d_ws, size_t ws_size,
                              hipStream_t stream) {
  (void)in_sizes; (void)n_in; (void)out_size; (void)ws_size;
  constexpr int NB = 4, L = 2048, E = 1024, H = 16, D = 64, OUT = 1024;
  constexpr int M = NB * L;        // 8192 tokens
  constexpr int N_QKV = H * 3 * D; // 3072

  const float* x      = (const float*)d_in[0];  // [NB,L,E]
  const float* Wa     = (const float*)d_in[1];  // [3072,1024]
  const float* Wout_w = (const float*)d_in[2];  // [1024,1024]
  const float* Wout_b = (const float*)d_in[3];  // [1024]
  float*       out    = (float*)d_out;          // [NB,L,OUT]

  // Workspace layout (u16 elements): ~88 MiB total.
  unsigned short* Xb   = (unsigned short*)d_ws;                 // 8192x1024
  unsigned short* Wab  = Xb   + (size_t)M * E;                  // 3072x1024
  unsigned short* Wob  = Wab  + (size_t)N_QKV * E;              // 1024x1024
  unsigned short* QKVb = Wob  + (size_t)OUT * (H * D);          // 8192x3072
  unsigned short* Obf  = QKVb + (size_t)M * N_QKV;              // 8192x1024

  // 1) Convert inputs/weights to bf16.
  convert_f32_to_bf16<<<2048, 256, 0, stream>>>(x, Xb, (size_t)M * E);
  convert_f32_to_bf16<<<1024, 256, 0, stream>>>(Wa, Wab, (size_t)N_QKV * E);
  convert_f32_to_bf16<<<512, 256, 0, stream>>>(Wout_w, Wob, (size_t)OUT * H * D);

  // 2) QKV projection: [8192,1024] x [3072,1024]^T -> bf16 [8192,3072].
  gemm_wmma_nt<true><<<dim3(N_QKV / 64, M / 128), 256, 0, stream>>>(
      Xb, Wab, QKVb, nullptr, nullptr, M, N_QKV, E);

  // 3) Flash attention -> bf16 O [8192,1024].
  flash_attn_wmma<<<dim3(L / 64, NB * H), 128, 0, stream>>>(QKVb, Obf);

  // 4) Output projection: [8192,1024] x [1024,1024]^T + bias -> f32 d_out.
  gemm_wmma_nt<false><<<dim3(OUT / 64, M / 128), 256, 0, stream>>>(
      Obf, Wob, nullptr, out, Wout_b, M, OUT, H * D);
}
